// CrossDeformableStateSpaceModel_87995289961154
// MI455X (gfx1250) — matrix operand order repaired and checked by reading, not compiled, not run on hardware
//
#include <hip/hip_runtime.h>
#include <math.h>

#define B_  8
#define CH  128
#define Hh  32
#define Ww  32
#define LL  1024
#define L2  2048
#define DIm 256
#define DSN 16
#define DTRm 8

typedef __attribute__((ext_vector_type(16))) _Float16 v16h;
typedef __attribute__((ext_vector_type(8)))  float    v8f;

__device__ __forceinline__ float silu_f(float x){ return x / (1.f + __expf(-x)); }

__device__ __forceinline__ float act_apply(float v, int act)
{
  if      (act == 1) v = v / (1.f + __expf(-v));
  else if (act == 2) v = (v > 20.f) ? v : log1pf(__expf(v));
  return v;
}

// ---------------------------------------------------------------------------
// Fast GEMM via V_WMMA_F32_16X16X32_F16:  C[M,N] = act(A[M,K] * W[N,K]^T + bias)
// REQUIRES: M % 16 == 0, K % 32 == 0, lda % 4 == 0, ldw % 4 == 0, 16B-aligned
// pointers. N may be ragged: weight-row index is clamped (always-in-bounds
// unconditional loads) and stores are masked.
// Fragment layouts per CDNA5 ISA 7.12.2:
//   A (16x32 f16): lane half selects K+0/ +8 (lo 16) and K+16 groups -> per lane
//   two contiguous 8-float runs -> 4x b128 loads.
//   B (32x16 f16): same K order as A -> per lane
//   two contiguous 8-float runs -> 4x b128 loads.
// ---------------------------------------------------------------------------
__global__ void gemm_wmma_fast(const float* __restrict__ A, int lda,
                               const float* __restrict__ W, int ldw,
                               const float* __restrict__ bias,
                               float* __restrict__ C, int ldc,
                               int M, int N, int K, int act)
{
  const int lane  = threadIdx.x & 31;
  const int wave  = threadIdx.x >> 5;
  const int mtile = (blockIdx.y * 4 + wave) << 4;
  const int ntile = blockIdx.x << 4;
  if (mtile >= M) return;                 // wave-uniform exit, EXEC stays full
  const int half = lane >> 4;
  const int l16  = lane & 15;
  const int arow = mtile + l16;
  const int wrow = ntile + l16;
  const int wrowc = (wrow < N) ? wrow : (N - 1);   // clamp: loads stay in-bounds

  const float4* __restrict__ Ar = (const float4*)(A + (long)arow  * lda);
  const float4* __restrict__ Wr = (const float4*)(W + (long)wrowc * ldw);

  v8f acc = {0.f,0.f,0.f,0.f,0.f,0.f,0.f,0.f};

  for (int k0 = 0; k0 < K; k0 += 32) {
    __builtin_prefetch(A + (long)arow * lda + k0 + 64, 0, 3);
    // A fragment: contiguous runs at k0 + half*8 and k0 + 16 + half*8
    const int ka = (k0 + (half << 3)) >> 2;
    float4 a01 = Ar[ka];     float4 a23 = Ar[ka + 1];
    float4 a45 = Ar[ka + 4]; float4 a67 = Ar[ka + 5];
    v16h a;
    a[0]=(_Float16)a01.x; a[1]=(_Float16)a01.y; a[2]=(_Float16)a01.z; a[3]=(_Float16)a01.w;
    a[4]=(_Float16)a23.x; a[5]=(_Float16)a23.y; a[6]=(_Float16)a23.z; a[7]=(_Float16)a23.w;
    a[8]=(_Float16)a45.x; a[9]=(_Float16)a45.y; a[10]=(_Float16)a45.z; a[11]=(_Float16)a45.w;
    a[12]=(_Float16)a67.x; a[13]=(_Float16)a67.y; a[14]=(_Float16)a67.z; a[15]=(_Float16)a67.w;
    // B fragment: contiguous runs at k0 + half*8 and k0 + 16 + half*8
    const int kb = (k0 + (half << 3)) >> 2;
    float4 b0 = Wr[kb];     float4 b1 = Wr[kb + 1];
    float4 b2 = Wr[kb + 4]; float4 b3 = Wr[kb + 5];
    v16h b;
    b[0]=(_Float16)b0.x; b[1]=(_Float16)b0.y; b[2]=(_Float16)b0.z; b[3]=(_Float16)b0.w;
    b[4]=(_Float16)b1.x; b[5]=(_Float16)b1.y; b[6]=(_Float16)b1.z; b[7]=(_Float16)b1.w;
    b[8]=(_Float16)b2.x; b[9]=(_Float16)b2.y; b[10]=(_Float16)b2.z; b[11]=(_Float16)b2.w;
    b[12]=(_Float16)b3.x; b[13]=(_Float16)b3.y; b[14]=(_Float16)b3.z; b[15]=(_Float16)b3.w;

    acc = __builtin_amdgcn_wmma_f32_16x16x32_f16(false, a, false, b,
                                                 (short)0, acc, false, false);
  }

  const int ncol = ntile + l16;
  if (ncol < N) {
    float bv = bias ? bias[ncol] : 0.f;
#pragma unroll
    for (int r = 0; r < 8; ++r) {
      int mrow = mtile + (half << 3) + r;     // C/D: lane half selects M, M+8
      C[(long)mrow * ldc + ncol] = act_apply(acc[r] + bv, act);
    }
  }
}

// ---------------------------------------------------------------------------
// Edge GEMM (used only for the dt projection: K=8, lda=40). Loads use clamped
// indices + value-side selects (no EXEC-mask branches, no per-element waits).
// Requires M % 16 == 0.
// ---------------------------------------------------------------------------
__global__ void gemm_wmma_edge(const float* __restrict__ A, int lda,
                               const float* __restrict__ W, int ldw,
                               const float* __restrict__ bias,
                               float* __restrict__ C, int ldc,
                               int M, int N, int K, int act)
{
  const int lane  = threadIdx.x & 31;
  const int wave  = threadIdx.x >> 5;
  const int mtile = (blockIdx.y * 4 + wave) << 4;
  const int ntile = blockIdx.x << 4;
  if (mtile >= M) return;
  const int half = lane >> 4;
  const int l16  = lane & 15;
  const int arow = mtile + l16;
  const int wrow  = ntile + l16;
  const int wrowc = (wrow < N) ? wrow : (N - 1);

  v8f acc = {0.f,0.f,0.f,0.f,0.f,0.f,0.f,0.f};

  for (int k0 = 0; k0 < K; k0 += 32) {
    v16h a, b;
#pragma unroll
    for (int i = 0; i < 8; ++i) {
      int kk  = k0 + ((i >> 2) << 4) + (half << 3) + ((i & 3) << 1);
      int kc0 = (kk     < K) ? kk     : (K - 1);
      int kc1 = (kk + 1 < K) ? kk + 1 : (K - 1);
      float a0 = A[(long)arow * lda + kc0];
      float a1 = A[(long)arow * lda + kc1];
      a[2*i]   = (_Float16)((kk     < K) ? a0 : 0.f);
      a[2*i+1] = (_Float16)((kk + 1 < K) ? a1 : 0.f);
    }
#pragma unroll
    for (int i = 0; i < 8; ++i) {
      int kk  = k0 + ((i >> 2) << 4) + (half << 3) + ((i & 3) << 1);
      int kc0 = (kk     < K) ? kk     : (K - 1);
      int kc1 = (kk + 1 < K) ? kk + 1 : (K - 1);
      float b0 = W[(long)wrowc * ldw + kc0];
      float b1 = W[(long)wrowc * ldw + kc1];
      b[2*i]   = (_Float16)((kk     < K && wrow < N) ? b0 : 0.f);
      b[2*i+1] = (_Float16)((kk + 1 < K && wrow < N) ? b1 : 0.f);
    }
    acc = __builtin_amdgcn_wmma_f32_16x16x32_f16(false, a, false, b,
                                                 (short)0, acc, false, false);
  }

  const int ncol = ntile + l16;
  if (ncol < N) {
    float bv = bias ? bias[ncol] : 0.f;
#pragma unroll
    for (int r = 0; r < 8; ++r) {
      int mrow = mtile + (half << 3) + r;
      C[(long)mrow * ldc + ncol] = act_apply(acc[r] + bv, act);
    }
  }
}

// ---------------------------------------------------------------------------
// Layout shuffles
// ---------------------------------------------------------------------------
__global__ void k_nchw_to_blc(const float* __restrict__ in, float* __restrict__ out, int n)
{
  int i = blockIdx.x * blockDim.x + threadIdx.x;
  if (i >= n) return;
  int l = i & (LL - 1); int c = (i >> 10) & (CH - 1); int b = i >> 17;
  out[((long)b * LL + l) * CH + c] = in[i];
}

__global__ void k_blc_to_nchw(const float* __restrict__ in, float* __restrict__ out, int n)
{
  int i = blockIdx.x * blockDim.x + threadIdx.x;
  if (i >= n) return;
  int l = i & (LL - 1); int c = (i >> 10) & (CH - 1); int b = i >> 17;
  out[i] = in[((long)b * LL + l) * CH + c];
}

// ---------------------------------------------------------------------------
// Depthwise 3x3 conv, SAME padding (cross-correlation), optional SiLU
// ---------------------------------------------------------------------------
__global__ void k_dwconv3x3(const float* __restrict__ in, const float* __restrict__ w,
                            const float* __restrict__ bias, float* __restrict__ out,
                            int do_silu)
{
  int i = blockIdx.x * blockDim.x + threadIdx.x;
  if (i >= B_ * CH * LL) return;
  int x = i & 31, y = (i >> 5) & 31, c = (i >> 10) & (CH - 1);
  const float* ip = in + ((long)(i >> 10) << 10);
  float s = bias[c];
#pragma unroll
  for (int ky = 0; ky < 3; ++ky) {
    int yy = y + ky - 1;
    if (yy < 0 || yy > 31) continue;
#pragma unroll
    for (int kx = 0; kx < 3; ++kx) {
      int xx = x + kx - 1;
      if (xx < 0 || xx > 31) continue;
      s += ip[yy * 32 + xx] * w[c * 9 + ky * 3 + kx];
    }
  }
  if (do_silu) s = silu_f(s);
  out[i] = s;
}

// Mean over spatial dim per (b,c)
__global__ void k_chan_mean(const float* __restrict__ in, float* __restrict__ out, int nbc)
{
  int i = blockIdx.x * blockDim.x + threadIdx.x;
  if (i >= nbc) return;
  const float* p = in + (long)i * LL;
  float s = 0.f;
  for (int l = 0; l < LL; ++l) s += p[l];
  out[i] = s * (1.f / LL);
}

// Channel-attention MLP: relu(128->8) then sigmoid(8->128), one block per batch
__global__ void k_ca_mlp(const float* __restrict__ yv,
                         const float* __restrict__ w1, const float* __restrict__ b1,
                         const float* __restrict__ w2, const float* __restrict__ b2,
                         float* __restrict__ yv2)
{
  __shared__ float sv[CH];
  __shared__ float t1[8];
  int b = blockIdx.x, t = threadIdx.x;
  sv[t] = yv[b * CH + t];
  __syncthreads();
  if (t < 8) {
    float s = b1[t];
    for (int k = 0; k < CH; ++k) s += w1[t * CH + k] * sv[k];
    t1[t] = fmaxf(s, 0.f);
  }
  __syncthreads();
  float s = b2[t];
#pragma unroll
  for (int k = 0; k < 8; ++k) s += w2[t * 8 + k] * t1[k];
  yv2[b * CH + t] = 1.f / (1.f + __expf(-s));
}

// x = gelu(x * yv2[b,c]) in place (exact gelu)
__global__ void k_scale_gelu(float* __restrict__ x, const float* __restrict__ yv2)
{
  int i = blockIdx.x * blockDim.x + threadIdx.x;
  if (i >= B_ * CH * LL) return;
  int c = (i >> 10) & (CH - 1); int b = i >> 17;
  float v = x[i] * yv2[b * CH + c];
  x[i] = 0.5f * v * (1.f + erff(v * 0.7071067811865476f));
}

// LayerNorm over channels at each (b,l). NCHW in/out. Block = 128 threads.
__global__ void k_ln_channel(const float* __restrict__ in, const float* __restrict__ g,
                             const float* __restrict__ bb, float eps, float* __restrict__ out)
{
  __shared__ float red[CH];
  int bl = blockIdx.x, c = threadIdx.x;
  int b = bl >> 10, l = bl & (LL - 1);
  long idx = ((long)(b * CH + c) << 10) + l;
  float v = in[idx];
  red[c] = v; __syncthreads();
  for (int s = 64; s > 0; s >>= 1) { if (c < s) red[c] += red[c + s]; __syncthreads(); }
  float mean = red[0] * (1.f / CH); __syncthreads();
  float d = v - mean;
  red[c] = d * d; __syncthreads();
  for (int s = 64; s > 0; s >>= 1) { if (c < s) red[c] += red[c + s]; __syncthreads(); }
  float var = red[0] * (1.f / CH);
  out[idx] = d * rsqrtf(var + eps) * g[c] + bb[c];
}

// off = tanh(off_w @ x); delta_p scaled by 1/W,1/H; di = ref + delta_t
__global__ void k_offset_proj(const float* __restrict__ x, const float* __restrict__ offw,
                              float* __restrict__ delta_p, float* __restrict__ di)
{
  int i = blockIdx.x * blockDim.x + threadIdx.x;
  if (i >= B_ * LL) return;
  int b = i >> 10, l = i & (LL - 1);
  const float* xp = x + (long)b * CH * LL + l;
  float s0 = 0.f, s1 = 0.f, s2 = 0.f;
  for (int c = 0; c < CH; ++c) {
    float v = xp[(long)c * LL];
    s0 += v * offw[c];
    s1 += v * offw[CH + c];
    s2 += v * offw[2 * CH + c];
  }
  delta_p[b * 2 * LL + l]      = tanhf(s0) * (1.f / 32.f);
  delta_p[b * 2 * LL + LL + l] = tanhf(s1) * (1.f / 32.f);
  float ref = (float)l * (2.f / (LL - 1)) - 1.f;
  di[i] = ref + tanhf(s2);
}

// Stable rank-based argsort, one block (1024 threads) per batch
__global__ void k_argsort(const float* __restrict__ di, int* __restrict__ order)
{
  __shared__ float d[LL];
  int b = blockIdx.x, t = threadIdx.x;
  d[t] = di[b * LL + t];
  __syncthreads();
  float mv = d[t];
  int r = 0;
  for (int j = 0; j < LL; ++j) {
    float dj = d[j];
    r += (dj < mv) || (dj == mv && j < t);
  }
  order[b * LL + r] = t;
}

// Bilinear grid-sample of x_dw and pan_dw at base grid + delta_p (zero-pad OOB)
__global__ void k_grid_sample(const float* __restrict__ xdw, const float* __restrict__ pandw,
                              const float* __restrict__ delta_p,
                              float* __restrict__ fx, float* __restrict__ fp)
{
  int i = blockIdx.x * blockDim.x + threadIdx.x;
  if (i >= B_ * CH * LL) return;
  int l = i & (LL - 1); int c = (i >> 10) & (CH - 1); int b = i >> 17;
  int px = l & 31, py = l >> 5;
  float gx = (px * (2.f / 31.f) - 1.f) + delta_p[b * 2 * LL + l];
  float gy = (py * (2.f / 31.f) - 1.f) + delta_p[b * 2 * LL + LL + l];
  float fxc = (gx + 1.f) * 0.5f * 32.f - 0.5f;
  float fyc = (gy + 1.f) * 0.5f * 32.f - 0.5f;
  float x0f = floorf(fxc), y0f = floorf(fyc);
  int x0 = (int)x0f, y0 = (int)y0f;
  float wx1 = fxc - x0f, wx0 = 1.f - wx1;
  float wy1 = fyc - y0f, wy0 = 1.f - wy1;
  const float* imx = xdw   + ((long)(b * CH + c) << 10);
  const float* imp = pandw + ((long)(b * CH + c) << 10);
  float ax = 0.f, ap = 0.f;
  auto tap = [&](int ix, int iy, float wgt) {
    if (ix >= 0 && ix < 32 && iy >= 0 && iy < 32) {
      int o = iy * 32 + ix;
      ax += imx[o] * wgt;
      ap += imp[o] * wgt;
    }
  };
  tap(x0,     y0,     wx0 * wy0);
  tap(x0 + 1, y0,     wx1 * wy0);
  tap(x0,     y0 + 1, wx0 * wy1);
  tap(x0 + 1, y0 + 1, wx1 * wy1);
  fx[i] = ax; fp[i] = ap;
}

// def_seq: gather_approx + interleave -> seq[b, 2l(+1), c]
__global__ void k_def_seq(const float* __restrict__ fx, const float* __restrict__ fp,
                          const float* __restrict__ fxm, const float* __restrict__ fpm,
                          const int* __restrict__ order, float* __restrict__ seq)
{
  int i = blockIdx.x * blockDim.x + threadIdx.x;
  if (i >= B_ * CH * LL) return;
  int l = i & (LL - 1); int c = (i >> 10) & (CH - 1); int b = i >> 17;
  int src = order[b * LL + l];
  long pbase = ((long)(b * CH + c) << 10);
  float mx = fxm[b * CH + c];
  float vx = (fx[pbase + src] - mx) + mx;     // matches (g - m) + m
  float mp = fpm[b * CH + c];
  float vp = (fp[pbase + src] - mp) + mp;
  long row = (long)b * L2 + 2 * l;
  seq[row * CH + c]        = vx;
  seq[(row + 1) * CH + c]  = vp;
}

// fwd/bwd interleaved sequences (flip = flip H)
__global__ void k_mk_seq(const float* __restrict__ x, const float* __restrict__ p,
                         float* __restrict__ seq, int flip)
{
  int i = blockIdx.x * blockDim.x + threadIdx.x;
  if (i >= B_ * CH * LL) return;
  int l = i & (LL - 1); int c = (i >> 10) & (CH - 1); int b = i >> 17;
  int src = flip ? (((31 - (l >> 5)) << 5) | (l & 31)) : l;
  long pbase = ((long)(b * CH + c) << 10);
  long row = (long)b * L2 + 2 * l;
  seq[row * CH + c]       = x[pbase + src];
  seq[(row + 1) * CH + c] = p[pbase + src];
}

// Causal depthwise conv1d (pad left 3) + SiLU; xc = first 256 cols of xz
__global__ void k_conv1d_silu(const float* __restrict__ xz, const float* __restrict__ w,
                              const float* __restrict__ bias, float* __restrict__ xs)
{
  int i = blockIdx.x * blockDim.x + threadIdx.x;
  if (i >= B_ * L2 * DIm) return;
  int d = i & (DIm - 1); int t = (i >> 8) & (L2 - 1); int b = i >> 19;
  long base = (long)b * L2 * 512;
  float s = 0.f;
#pragma unroll
  for (int j = 0; j < 4; ++j) {
    int tt = t - 3 + j;
    if (tt >= 0) s += xz[base + (long)tt * 512 + d] * w[d * 4 + j];
  }
  s += bias[d];
  xs[i] = silu_f(s);
}

// Selective scan: 1 thread per (b,d), 16-state recurrence in registers.
// y output may alias dt (read-before-write per element).
__global__ void k_scan(const float* __restrict__ xs, const float* __restrict__ dtb,
                       const float* __restrict__ dbl, const float* __restrict__ xz,
                       const float* __restrict__ A_log, const float* __restrict__ Dp,
                       float* __restrict__ y)
{
  int b = blockIdx.x, d = threadIdx.x;
  float Ar[DSN];
#pragma unroll
  for (int n = 0; n < DSN; ++n) Ar[n] = -__expf(A_log[d * DSN + n]);
  float Dd = Dp[d];
  float h[DSN];
#pragma unroll
  for (int n = 0; n < DSN; ++n) h[n] = 0.f;

  for (int t = 0; t < L2; ++t) {
    long row = (long)b * L2 + t;
    float dtv = dtb[row * DIm + d];
    float xv  = xs[row * DIm + d];
    const float* Bt = dbl + row * 40 + DTRm;
    const float* Ct = dbl + row * 40 + DTRm + DSN;
    float dx = dtv * xv;
    float acc = 0.f;
#pragma unroll
    for (int n = 0; n < DSN; ++n) {
      h[n] = __expf(dtv * Ar[n]) * h[n] + dx * Bt[n];
      acc += h[n] * Ct[n];
    }
    acc += xv * Dd;
    float z = xz[row * 512 + DIm + d];
    acc *= silu_f(z);
    y[row * DIm + d] = acc;
  }
}

// Fused merge + gate + final LayerNorm over channels, output NCHW
__global__ void k_final(const float* __restrict__ f, const float* __restrict__ bo,
                        const float* __restrict__ dd, const float* __restrict__ gate,
                        const float* __restrict__ g, const float* __restrict__ bb,
                        float* __restrict__ out)
{
  __shared__ float red[CH];
  int bl = blockIdx.x, c = threadIdx.x;
  int b = bl >> 10, l = bl & (LL - 1);
  long m = (long)bl;
  float v = (f[m * CH + c] + bo[m * CH + c] + dd[m * CH + c]) * (1.f / 3.f);
  v *= gate[((long)(b * CH + c) << 10) + l];
  red[c] = v; __syncthreads();
  for (int s = 64; s > 0; s >>= 1) { if (c < s) red[c] += red[c + s]; __syncthreads(); }
  float mean = red[0] * (1.f / CH); __syncthreads();
  float dv = v - mean;
  red[c] = dv * dv; __syncthreads();
  for (int s = 64; s > 0; s >>= 1) { if (c < s) red[c] += red[c + s]; __syncthreads(); }
  float var = red[0] * (1.f / CH);
  out[((long)(b * CH + c) << 10) + l] = dv * rsqrtf(var + 1e-5f) * g[c] + bb[c];
}

// ---------------------------------------------------------------------------
extern "C" void kernel_launch(void* const* d_in, const int* in_sizes, int n_in,
                              void* d_out, int out_size, void* d_ws, size_t ws_size,
                              hipStream_t stream)
{
  (void)in_sizes; (void)n_in; (void)out_size;
  auto P = [&](int i) { return (const float*)d_in[i]; };

  const float* ms       = P(0);
  const float* pan      = P(1);
  const float* pre_w    = P(2);
  const float* pre_b    = P(3);
  const float* dwc_w    = P(4);
  const float* dwc_b    = P(5);
  const float* off_dw_w = P(6);
  const float* off_dw_b = P(7);
  const float* ca1_w    = P(8);
  const float* ca1_b    = P(9);
  const float* ca2_w    = P(10);
  const float* ca2_b    = P(11);
  const float* off_ln_g = P(12);
  const float* off_ln_b = P(13);
  const float* off_w    = P(14);
  // fwd_ssm base 15, bwd_ssm base 24:
  //   +0 in_proj +1 conv_w +2 conv_b +3 x_proj +4 dt_w +5 dt_b +6 A_log +7 D +8 out_proj
  const float* fwd_cmp_w = P(33); const float* fwd_cmp_b = P(34);
  const float* bwd_cmp_w = P(35); const float* bwd_cmp_b = P(36);
  const float* def_cmp_w = P(37); const float* def_cmp_b = P(38);
  const float* gate_w    = P(39); const float* gate_b    = P(40);
  const float* ln_g      = P(41); const float* ln_b      = P(42);

  float* ws = (float*)d_ws;
  const size_t NCL = (size_t)B_ * CH * LL;      // 1,048,576
  const size_t SEQ = (size_t)B_ * L2 * CH;      // 2,097,152
  size_t o = 0;
  auto alloc = [&](size_t n) { size_t r = o; o += n; return r; };
  const size_t o_mslin = alloc(NCL);            // also offset-net feature reuse
  const size_t o_xdw   = alloc(NCL);
  const size_t o_pandw = alloc(NCL);
  const size_t o_fx    = alloc(NCL);            // fx+fp reused as mamba_out (2*NCL)
  const size_t o_fp    = alloc(NCL);
  const size_t o_fout  = alloc(NCL);
  const size_t o_bout  = alloc(NCL);
  const size_t o_dcmp  = alloc(NCL);
  const size_t o_gate  = alloc(NCL);
  const size_t o_seq   = alloc(SEQ);
  const size_t o_xz    = alloc((size_t)B_ * L2 * 512);
  const size_t o_xs    = alloc((size_t)B_ * L2 * DIm);
  const size_t o_dbl   = alloc((size_t)B_ * L2 * 40);
  const size_t o_dt    = alloc((size_t)B_ * L2 * DIm);   // scan y aliases this
  const size_t o_yvm   = alloc(B_ * CH);
  const size_t o_yv2   = alloc(B_ * CH);
  const size_t o_dp    = alloc(B_ * 2 * LL);
  const size_t o_di    = alloc(B_ * LL);
  const size_t o_fxm   = alloc(B_ * CH);
  const size_t o_fpm   = alloc(B_ * CH);
  const size_t o_order = alloc(B_ * LL);        // ints
  if (ws_size < o * sizeof(float)) return;      // workspace too small: bail

  (void)o_fp;
  auto nb = [](long n) { return dim3((unsigned)((n + 255) / 256)); };
  auto gemm = [&](const float* A, int lda, const float* W, int ldw, const float* bias,
                  float* C, int ldc, int M, int N, int K, int act) {
    dim3 g((N + 15) / 16, (M + 63) / 64);
    gemm_wmma_fast<<<g, 128, 0, stream>>>(A, lda, W, ldw, bias, C, ldc, M, N, K, act);
  };
  auto gemm_edge = [&](const float* A, int lda, const float* W, int ldw, const float* bias,
                       float* C, int ldc, int M, int N, int K, int act) {
    dim3 g((N + 15) / 16, (M + 63) / 64);
    gemm_wmma_edge<<<g, 128, 0, stream>>>(A, lda, W, ldw, bias, C, ldc, M, N, K, act);
  };
  const int nNCL = (int)NCL;

  // ---- 1. pre-linear (WMMA) ----
  k_nchw_to_blc<<<nb(nNCL), 256, 0, stream>>>(ms, ws + o_seq, nNCL);
  gemm(ws + o_seq, CH, pre_w, CH, pre_b, ws + o_seq + NCL, CH, B_ * LL, CH, CH, 0);
  k_blc_to_nchw<<<nb(nNCL), 256, 0, stream>>>(ws + o_seq + NCL, ws + o_mslin, nNCL);

  // ---- 2. depthwise conv + SiLU ----
  k_dwconv3x3<<<nb(nNCL), 256, 0, stream>>>(ws + o_mslin, dwc_w, dwc_b, ws + o_xdw, 1);
  k_dwconv3x3<<<nb(nNCL), 256, 0, stream>>>(pan, dwc_w, dwc_b, ws + o_pandw, 1);

  // ---- 3. offset network ----
  k_dwconv3x3<<<nb(nNCL), 256, 0, stream>>>(ws + o_xdw, off_dw_w, off_dw_b, ws + o_mslin, 0);
  k_chan_mean<<<nb(B_ * CH), 256, 0, stream>>>(ws + o_mslin, ws + o_yvm, B_ * CH);
  k_ca_mlp<<<B_, CH, 0, stream>>>(ws + o_yvm, ca1_w, ca1_b, ca2_w, ca2_b, ws + o_yv2);
  k_scale_gelu<<<nb(nNCL), 256, 0, stream>>>(ws + o_mslin, ws + o_yv2);
  k_ln_channel<<<B_ * LL, CH, 0, stream>>>(ws + o_mslin, off_ln_g, off_ln_b, 1e-6f, ws + o_mslin);
  k_offset_proj<<<nb(B_ * LL), 256, 0, stream>>>(ws + o_mslin, off_w, ws + o_dp, ws + o_di);

  // ---- 4. deformable scan ----
  k_argsort<<<B_, LL, 0, stream>>>(ws + o_di, (int*)(ws + o_order));
  k_grid_sample<<<nb(nNCL), 256, 0, stream>>>(ws + o_xdw, ws + o_pandw, ws + o_dp,
                                              ws + o_fx, ws + o_fp);
  k_chan_mean<<<nb(B_ * CH), 256, 0, stream>>>(ws + o_fx, ws + o_fxm, B_ * CH);
  k_chan_mean<<<nb(B_ * CH), 256, 0, stream>>>(ws + o_fp, ws + o_fpm, B_ * CH);
  k_def_seq<<<nb(nNCL), 256, 0, stream>>>(ws + o_fx, ws + o_fp, ws + o_fxm, ws + o_fpm,
                                          (const int*)(ws + o_order), ws + o_seq);
  // def compressor: seq (b,2L,128) reinterpreted as (b*L, 256)
  gemm(ws + o_seq, 2 * CH, def_cmp_w, 2 * CH, def_cmp_b, ws + o_dcmp, CH,
       B_ * LL, CH, 2 * CH, 1);

  // ---- 5. Mamba branches ----
  auto run_mamba = [&](int pb, const float* seqp, float* outp) {
    const float* in_proj = P(pb + 0); const float* conv_w = P(pb + 1);
    const float* conv_b  = P(pb + 2); const float* x_proj = P(pb + 3);
    const float* dt_w    = P(pb + 4); const float* dt_b   = P(pb + 5);
    const float* A_log   = P(pb + 6); const float* Dp     = P(pb + 7);
    const float* out_prj = P(pb + 8);
    gemm(seqp, CH, in_proj, CH, nullptr, ws + o_xz, 512, B_ * L2, 512, CH, 0);
    k_conv1d_silu<<<nb((long)B_ * L2 * DIm), 256, 0, stream>>>(ws + o_xz, conv_w, conv_b,
                                                               ws + o_xs);
    gemm(ws + o_xs, DIm, x_proj, DIm, nullptr, ws + o_dbl, 40, B_ * L2, 40, DIm, 0);
    gemm_edge(ws + o_dbl, 40, dt_w, DTRm, dt_b, ws + o_dt, DIm, B_ * L2, DIm, DTRm, 2);
    k_scan<<<B_, DIm, 0, stream>>>(ws + o_xs, ws + o_dt, ws + o_dbl, ws + o_xz,
                                   A_log, Dp, ws + o_dt);
    gemm(ws + o_dt, DIm, out_prj, DIm, nullptr, outp, CH, B_ * L2, CH, DIm, 0);
  };

  k_mk_seq<<<nb(nNCL), 256, 0, stream>>>(ws + o_xdw, ws + o_pandw, ws + o_seq, 0);
  run_mamba(15, ws + o_seq, ws + o_fx);
  gemm(ws + o_fx, 2 * CH, fwd_cmp_w, 2 * CH, fwd_cmp_b, ws + o_fout, CH,
       B_ * LL, CH, 2 * CH, 1);

  k_mk_seq<<<nb(nNCL), 256, 0, stream>>>(ws + o_xdw, ws + o_pandw, ws + o_seq, 1);
  run_mamba(24, ws + o_seq, ws + o_fx);
  gemm(ws + o_fx, 2 * CH, bwd_cmp_w, 2 * CH, bwd_cmp_b, ws + o_bout, CH,
       B_ * LL, CH, 2 * CH, 1);

  // ---- 6. gate GEMM along W (WMMA, M=b*c*h=32768, N=K=32) ----
  gemm(ms, 32, gate_w, 32, gate_b, ws + o_gate, 32, B_ * CH * Hh, 32, 32, 1);

  // ---- 7. fused merge + gate + LayerNorm -> output NCHW ----
  k_final<<<B_ * LL, CH, 0, stream>>>(ws + o_fout, ws + o_bout, ws + o_dcmp,
                                      ws + o_gate, ln_g, ln_b, (float*)d_out);
}
